// GraphEncoder_57475252355630
// MI455X (gfx1250) — matrix operand order repaired and checked by reading, not compiled, not run on hardware
//
#include <hip/hip_runtime.h>
#include <hip/hip_bf16.h>
#include <math.h>

// ---------------- problem constants (from reference) ----------------
#define NN    65536      // nodes
#define NE    524288     // real edges
#define NEF   589824     // edges + self loops
#define DD    128        // hidden dim
#define NHD   8          // heads
#define CC    16         // channels/head
#define EDIM  64         // edge feature dim
#define NPGC  1024       // nodes per graph
#define NBG   64         // graphs
#define KTOP  256        // top-k
#define NLAY  3

typedef unsigned short u16;
typedef __attribute__((ext_vector_type(16))) __bf16 v16bf;
typedef __attribute__((ext_vector_type(8)))  float  v8f;

// ---------------- small device helpers ----------------
static __device__ __forceinline__ u16 f32_to_bf16(float f) {
  unsigned u = __float_as_uint(f);
  if ((u & 0x7F800000u) == 0x7F800000u) return (u16)(u >> 16);   // inf/nan
  unsigned r = u + 0x7FFFu + ((u >> 16) & 1u);                   // RNE
  return (u16)(r >> 16);
}
// order-preserving float<->int transform (involution) for atomicMax on floats
static __device__ __forceinline__ int f2ord(float f) {
  int i = __float_as_int(f); return (i >= 0) ? i : (i ^ 0x7FFFFFFF);
}
static __device__ __forceinline__ float ord2f(int i) {
  return __int_as_float((i >= 0) ? i : (i ^ 0x7FFFFFFF));
}
#define NEG_ORD ((int)0x807FFFFF)   // f2ord(-inf)

static __device__ __forceinline__ void edge_sd(int e, const int* __restrict__ src,
                                               const int* __restrict__ dst, int& s, int& d) {
  if (e < NE) { s = src[e]; d = dst[e]; } else { s = e - NE; d = s; }  // self-loop tail
}

// ---------------- generic fills ----------------
__global__ void ge_fill_f32(float* p, int n, float v) {
  int t = blockIdx.x * blockDim.x + threadIdx.x;
  if (t < n) p[t] = v;
}
__global__ void ge_fill_i32(int* p, int n, int v) {
  int t = blockIdx.x * blockDim.x + threadIdx.x;
  if (t < n) p[t] = v;
}

// ---------------- input embedding: sinusoidal PE + node type embedding ----------------
__global__ void ge_pe_embed(const int* __restrict__ label, const float* __restrict__ node_emb,
                            float* __restrict__ x) {
  int t = blockIdx.x * blockDim.x + threadIdx.x;           // (node, pair j)
  if (t >= NN * (DD / 2)) return;
  int i = t >> 6, j = t & 63;
  float pos = (float)(i & (NPGC - 1));                     // graphs contiguous, equal size
  float div = __expf(-(float)j * (9.210340371976184f / (float)DD));  // ln(1e4)
  float ang = pos * div;
  const float* eb = node_emb + (size_t)label[i] * DD;
  float* xr = x + (size_t)i * DD;
  xr[2 * j]     = __sinf(ang) + eb[2 * j];
  xr[2 * j + 1] = __cosf(ang) + eb[2 * j + 1];
}

// ---------------- self-loop edge attr = mean of incoming edge attrs ----------------
__global__ void ge_loop_attr_acc(const int* __restrict__ dst, const int* __restrict__ elab,
                                 const float* __restrict__ eemb, float* __restrict__ lacc,
                                 float* __restrict__ deg) {
  int t = blockIdx.x * blockDim.x + threadIdx.x;           // (edge, 16-ch chunk)
  if (t >= NE * 4) return;
  int e = t >> 2, c0 = (t & 3) * 16;
  int d = dst[e];
  const float* s = eemb + (size_t)elab[e] * EDIM + c0;
  float* o = lacc + (size_t)d * EDIM + c0;
#pragma unroll
  for (int c = 0; c < 16; ++c) atomicAdd(&o[c], s[c]);
  if (c0 == 0) atomicAdd(&deg[d], 1.0f);
}
__global__ void ge_loop_attr_div(float* __restrict__ lacc, const float* __restrict__ deg) {
  int t = blockIdx.x * blockDim.x + threadIdx.x;
  if (t >= NN * EDIM) return;
  lacc[t] /= fmaxf(deg[t >> 6], 1.0f);
}

// ---------------- WMMA fragment packing (bf16) ----------------
// A fragment (16x32 MxK, 16-bit): lane L holds row M=L&15;
// lanes 0-15: K = 0..15; lanes 16-31: K = 16..31 (matches B packing)
__global__ void ge_pack_a(const float* __restrict__ src, u16* __restrict__ dst,
                          int Mtiles, int K) {
  int K32 = K >> 5;
  int t = blockIdx.x * blockDim.x + threadIdx.x;           // (mtile, kt, lane)
  if (t >= Mtiles * K32 * 32) return;
  int lane = t & 31, kt = (t >> 5) % K32, mt = (t >> 5) / K32;
  int row = mt * 16 + (lane & 15);
  int kb  = kt * 32 + ((lane < 16) ? 0 : 16);
  const float* s = src + (size_t)row * K;
  u16* d = dst + (size_t)t * 16;
#pragma unroll
  for (int j = 0; j < 8; ++j) d[j]     = f32_to_bf16(s[kb + j]);
#pragma unroll
  for (int j = 0; j < 8; ++j) d[8 + j] = f32_to_bf16(s[kb + 8 + j]);
}
// B fragment (32x16 KxN): lane L holds col N=L&15; lanes 0-15: K=0..15; 16-31: K=16..31
__global__ void ge_pack_b(const float* __restrict__ W, u16* __restrict__ dst,
                          int Ntiles, int K, int ldn) {
  int K32 = K >> 5;
  int t = blockIdx.x * blockDim.x + threadIdx.x;           // (ntile, kt, lane)
  if (t >= Ntiles * K32 * 32) return;
  int lane = t & 31, kt = (t >> 5) % K32, nt = (t >> 5) / K32;
  int col = nt * 16 + (lane & 15);
  int kb  = kt * 32 + ((lane < 16) ? 0 : 16);
  u16* d = dst + (size_t)t * 16;
#pragma unroll
  for (int j = 0; j < 16; ++j) d[j] = f32_to_bf16(W[(size_t)(kb + j) * ldn + col]);
}

// ---------------- bf16 WMMA GEMM: C[M x N] = Apk @ Bpk (+bias) ----------------
template <int K32>
__global__ void ge_gemm(const u16* __restrict__ Apk, const u16* __restrict__ Bpk,
                        const float* __restrict__ bias, float* __restrict__ C,
                        int ldc, int Ntiles, int nWaves) {
  int wid  = (blockIdx.x * blockDim.x + threadIdx.x) >> 5;
  int lane = threadIdx.x & 31;
  if (wid >= nWaves) return;                               // wave-uniform: EXEC stays full
  int mt = wid / Ntiles, nt = wid % Ntiles;
  const v16bf* Ap = reinterpret_cast<const v16bf*>(Apk + ((size_t)mt * K32 * 32 + lane) * 16);
  const v16bf* Bp = reinterpret_cast<const v16bf*>(Bpk + ((size_t)nt * K32 * 32 + lane) * 16);
  v8f acc = {0.f, 0.f, 0.f, 0.f, 0.f, 0.f, 0.f, 0.f};
#pragma unroll
  for (int kt = 0; kt < K32; ++kt) {
    v16bf a = Ap[(size_t)kt * 32];
    v16bf b = Bp[(size_t)kt * 32];
    acc = __builtin_amdgcn_wmma_f32_16x16x32_bf16(false, a, false, b, (short)0, acc,
                                                  false, false);
  }
  // C/D layout: VGPR r: lanes 0-15 -> M=r, lanes 16-31 -> M=8+r; N = lane&15
  int col   = nt * 16 + (lane & 15);
  float bv  = bias ? bias[col] : 0.0f;
  int rbase = (lane < 16) ? 0 : 8;
#pragma unroll
  for (int r = 0; r < 8; ++r)
    C[(size_t)(mt * 16 + rbase + r) * ldc + col] = acc[r] + bv;
}

// ---------------- GAT pieces ----------------
__global__ void ge_init_bias_rows(float* __restrict__ out, const float* __restrict__ bias) {
  int t = blockIdx.x * blockDim.x + threadIdx.x;
  if (t < NN * DD) out[t] = bias[t & (DD - 1)];
}
// weae[k][h] = sum_c We[k, h*16+c] * ae[h,c]   (64x8, per layer)
__global__ void ge_make_weae(const float* __restrict__ We, const float* __restrict__ ae,
                             float* __restrict__ out) {
  int t = blockIdx.x * blockDim.x + threadIdx.x;
  if (t >= EDIM * NHD) return;
  int k = t >> 3, h = t & 7;
  float acc = 0.f;
#pragma unroll
  for (int c = 0; c < CC; ++c) acc += We[(size_t)k * DD + h * CC + c] * ae[h * CC + c];
  out[t] = acc;
}
// per-node attention terms: alpha_s/alpha_d = sum_c h[i,h,c]*a{s,d}[h,c]
__global__ void ge_alpha_nodes(const float* __restrict__ hfeat, const float* __restrict__ asv,
                               const float* __restrict__ adv, float* __restrict__ as_n,
                               float* __restrict__ ad_n) {
  int t = blockIdx.x * blockDim.x + threadIdx.x;
  if (t >= NN * NHD) return;
  int i = t >> 3, h = t & 7;
  const float* hr = hfeat + (size_t)i * DD + h * CC;
  float sa = 0.f, sd = 0.f;
#pragma unroll
  for (int c = 0; c < CC; ++c) { sa += hr[c] * asv[h * CC + c]; sd += hr[c] * adv[h * CC + c]; }
  as_n[t] = sa; ad_n[t] = sd;
}
// per-edge raw score (incl. edge-feature term), leaky-relu, and segment max (atomic)
__global__ void ge_edge_score(const int* __restrict__ src, const int* __restrict__ dst,
                              const int* __restrict__ elab, const float* __restrict__ eemb,
                              const float* __restrict__ lacc, const float* __restrict__ weae,
                              const float* __restrict__ as_n, const float* __restrict__ ad_n,
                              float* __restrict__ abuf, int* __restrict__ mord) {
  int e = blockIdx.x * blockDim.x + threadIdx.x;
  if (e >= NEF) return;
  int s, d; edge_sd(e, src, dst, s, d);
  const float* ea = (e < NE) ? (eemb + (size_t)elab[e] * EDIM) : (lacc + (size_t)s * EDIM);
  float av[NHD];
#pragma unroll
  for (int h = 0; h < NHD; ++h) av[h] = 0.f;
  for (int k = 0; k < EDIM; ++k) {
    float v = ea[k];
    const float* w = weae + k * NHD;
#pragma unroll
    for (int h = 0; h < NHD; ++h) av[h] += v * w[h];
  }
#pragma unroll
  for (int h = 0; h < NHD; ++h) {
    float a = as_n[s * NHD + h] + ad_n[d * NHD + h] + av[h];
    a = (a > 0.f) ? a : 0.2f * a;                          // leaky relu
    abuf[(size_t)e * NHD + h] = a;
    atomicMax(&mord[d * NHD + h], f2ord(a));
  }
}
__global__ void ge_edge_exp(const int* __restrict__ src, const int* __restrict__ dst,
                            float* __restrict__ abuf, const int* __restrict__ mord,
                            float* __restrict__ zbuf) {
  int t = blockIdx.x * blockDim.x + threadIdx.x;
  if (t >= NEF * NHD) return;
  int e = t >> 3, h = t & 7;
  int s, d; edge_sd(e, src, dst, s, d); (void)s;
  float a = __expf(abuf[t] - ord2f(mord[d * NHD + h]));
  abuf[t] = a;
  atomicAdd(&zbuf[d * NHD + h], a);
}
__global__ void ge_edge_agg(const int* __restrict__ src, const int* __restrict__ dst,
                            const float* __restrict__ hfeat, const float* __restrict__ abuf,
                            const float* __restrict__ zbuf, float* __restrict__ out) {
  int t = blockIdx.x * blockDim.x + threadIdx.x;           // (edge, head)
  if (t >= NEF * NHD) return;
  int e = t >> 3, h = t & 7;
  int s, d; edge_sd(e, src, dst, s, d);
  float w = abuf[t] / zbuf[d * NHD + h];
  const float* hr = hfeat + (size_t)s * DD + h * CC;
  float* orow = out + (size_t)d * DD + h * CC;
#pragma unroll
  for (int c = 0; c < CC; ++c) atomicAdd(&orow[c], hr[c] * w);
}

// ---------------- graph LayerNorm (mode='graph') + relu + residual ----------------
__global__ void ge_ln_stats(const float* __restrict__ out, float* __restrict__ stats) {
  __shared__ float s1[256], s2[256];
  int g = blockIdx.x, tid = threadIdx.x;
  const float* p = out + (size_t)g * NPGC * DD;
  float s = 0.f, q = 0.f;
  for (int i = tid; i < NPGC * DD; i += 256) { float v = p[i]; s += v; q += v * v; }
  s1[tid] = s; s2[tid] = q; __syncthreads();
  for (int st = 128; st > 0; st >>= 1) {
    if (tid < st) { s1[tid] += s1[tid + st]; s2[tid] += s2[tid + st]; }
    __syncthreads();
  }
  if (tid == 0) {
    float inv = 1.0f / (float)(NPGC * DD);
    float mean = s1[0] * inv;
    float var  = fmaxf(s2[0] * inv - mean * mean, 0.f);
    stats[g * 2]     = mean;
    stats[g * 2 + 1] = rsqrtf(var + 1e-5f);
  }
}
__global__ void ge_ln_apply(const float* __restrict__ out, const float* __restrict__ stats,
                            const float* __restrict__ lnw, const float* __restrict__ lnb,
                            float* __restrict__ x) {
  int t = blockIdx.x * blockDim.x + threadIdx.x;
  if (t >= NN * DD) return;
  int dch = t & (DD - 1);
  int g = t >> 17;                                         // node>>10: (t>>7)>>10
  float v = (out[t] - stats[g * 2]) * stats[g * 2 + 1] * lnw[dch] + lnb[dch];
  x[t] = fmaxf(v, 0.f) + x[t];                             // relu + residual
}

// ---------------- ASAPooling ----------------
__global__ void ge_pool_max(const int* __restrict__ src, const int* __restrict__ dst,
                            const float* __restrict__ x, int* __restrict__ xqord) {
  int t = blockIdx.x * blockDim.x + threadIdx.x;           // (edge, 16-ch chunk)
  if (t >= NEF * NHD) return;
  int e = t >> 3, h = t & 7;
  int s, d; edge_sd(e, src, dst, s, d);
  const float* xr = x + (size_t)s * DD + h * CC;
  int* o = xqord + (size_t)d * DD + h * CC;
#pragma unroll
  for (int c = 0; c < CC; ++c) atomicMax(&o[c], f2ord(xr[c]));
}
__global__ void ge_ord2f(const int* __restrict__ a, float* __restrict__ b, int n) {
  int t = blockIdx.x * blockDim.x + threadIdx.x;
  if (t < n) b[t] = ord2f(a[t]);
}
__global__ void ge_pool_q(const float* __restrict__ xql, const float* __restrict__ x,
                          const float* __restrict__ attw, float* __restrict__ q1,
                          float* __restrict__ q2) {
  int i = blockIdx.x * blockDim.x + threadIdx.x;
  if (i >= NN) return;
  const float* a = xql + (size_t)i * DD;
  const float* b = x + (size_t)i * DD;
  float r1 = 0.f, r2 = 0.f;
#pragma unroll 8
  for (int c = 0; c < DD; ++c) { r1 += a[c] * attw[c]; r2 += b[c] * attw[DD + c]; }
  q1[i] = r1; q2[i] = r2;
}
__global__ void ge_pool_score(const int* __restrict__ src, const int* __restrict__ dst,
                              const float* __restrict__ q1, const float* __restrict__ q2,
                              const float* __restrict__ attb, float* __restrict__ sbuf,
                              int* __restrict__ smord) {
  int e = blockIdx.x * blockDim.x + threadIdx.x;
  if (e >= NEF) return;
  int s, d; edge_sd(e, src, dst, s, d);
  float sc = q1[d] + q2[s] + attb[0];
  sc = (sc > 0.f) ? sc : 0.2f * sc;
  sbuf[e] = sc;
  atomicMax(&smord[d], f2ord(sc));
}
__global__ void ge_pool_exp(const int* __restrict__ src, const int* __restrict__ dst,
                            float* __restrict__ sbuf, const int* __restrict__ smord,
                            float* __restrict__ sz) {
  int e = blockIdx.x * blockDim.x + threadIdx.x;
  if (e >= NEF) return;
  int s, d; edge_sd(e, src, dst, s, d); (void)s;
  float v = __expf(sbuf[e] - ord2f(smord[d]));
  sbuf[e] = v;
  atomicAdd(&sz[d], v);
}
__global__ void ge_pool_agg(const int* __restrict__ src, const int* __restrict__ dst,
                            const float* __restrict__ x, const float* __restrict__ sbuf,
                            const float* __restrict__ sz, float* __restrict__ xnew) {
  int t = blockIdx.x * blockDim.x + threadIdx.x;           // (edge, 16-ch chunk)
  if (t >= NEF * NHD) return;
  int e = t >> 3, h = t & 7;
  int s, d; edge_sd(e, src, dst, s, d);
  float w = sbuf[e] / sz[d];
  const float* xr = x + (size_t)s * DD + h * CC;
  float* o = xnew + (size_t)d * DD + h * CC;
#pragma unroll
  for (int c = 0; c < CC; ++c) atomicAdd(&o[c], xr[c] * w);
}

// ---------------- LEConv fitness + sigmoid ----------------
__global__ void ge_fit_node(const float* __restrict__ xnew, const float* __restrict__ w1,
                            const float* __restrict__ b1, const float* __restrict__ w2,
                            const float* __restrict__ w3, float* __restrict__ fa,
                            float* __restrict__ fb, float* __restrict__ ft) {
  int i = blockIdx.x * blockDim.x + threadIdx.x;
  if (i >= NN) return;
  const float* xr = xnew + (size_t)i * DD;
  float r1 = 0.f, r2 = 0.f, r3 = 0.f;
#pragma unroll 8
  for (int c = 0; c < DD; ++c) { float v = xr[c]; r1 += v * w1[c]; r2 += v * w2[c]; r3 += v * w3[c]; }
  fa[i] = r1 + b1[0]; fb[i] = r2; ft[i] = r3;
}
__global__ void ge_fit_edge(const int* __restrict__ src, const int* __restrict__ dst,
                            const float* __restrict__ fa, const float* __restrict__ fb,
                            float* __restrict__ msg) {
  int e = blockIdx.x * blockDim.x + threadIdx.x;
  if (e >= NEF) return;
  int s, d; edge_sd(e, src, dst, s, d);
  atomicAdd(&msg[d], fa[s] - fb[d]);
}
__global__ void ge_fit_final(const float* __restrict__ msg, const float* __restrict__ ft,
                             const float* __restrict__ b3, float* __restrict__ fit) {
  int i = blockIdx.x * blockDim.x + threadIdx.x;
  if (i >= NN) return;
  float v = msg[i] + ft[i] + b3[0];
  fit[i] = 1.0f / (1.0f + __expf(-v));
}

// ---------------- per-graph top-k via LDS bitonic sort (desc, idx tiebreak) ----------------
__global__ void ge_topk(const float* __restrict__ fit, float* __restrict__ vals,
                        int* __restrict__ perm) {
  __shared__ float sv[NPGC];
  __shared__ int   si[NPGC];
  int g = blockIdx.x, t = threadIdx.x;
  sv[t] = fit[(size_t)g * NPGC + t]; si[t] = t;
  __syncthreads();
  for (int k = 2; k <= NPGC; k <<= 1) {
    for (int j = k >> 1; j > 0; j >>= 1) {
      int ixj = t ^ j;
      if (ixj > t) {
        float v1 = sv[t], v2 = sv[ixj]; int i1 = si[t], i2 = si[ixj];
        bool after = (v1 < v2) || (v1 == v2 && i1 > i2);   // t ranks after ixj (descending)
        bool dir = ((t & k) == 0);
        if (after == dir) { sv[t] = v2; sv[ixj] = v1; si[t] = i2; si[ixj] = i1; }
      }
      __syncthreads();
    }
  }
  if (t < KTOP) {
    vals[g * KTOP + t] = sv[t];
    perm[g * KTOP + t] = g * NPGC + si[t];
  }
}
__global__ void ge_gather_out(const float* __restrict__ xnew, const int* __restrict__ perm,
                              const float* __restrict__ vals, float* __restrict__ out) {
  int t = blockIdx.x * blockDim.x + threadIdx.x;
  if (t >= NBG * KTOP * DD) return;
  int r = t >> 7, c = t & (DD - 1);
  out[t] = xnew[(size_t)perm[r] * DD + c] * vals[r];
}
__global__ void ge_batch_out(float* __restrict__ out) {
  int t = blockIdx.x * blockDim.x + threadIdx.x;
  if (t < NBG * KTOP) out[(size_t)NBG * KTOP * DD + t] = (float)(t >> 8);  // t / KTOP
}

// ---------------- host launcher ----------------
static inline int blks(long long n, int b) { return (int)((n + b - 1) / b); }

extern "C" void kernel_launch(void* const* d_in, const int* in_sizes, int n_in,
                              void* d_out, int out_size, void* d_ws, size_t ws_size,
                              hipStream_t stream) {
  (void)in_sizes; (void)n_in; (void)out_size; (void)ws_size;
  const int*   label    = (const int*)d_in[0];
  const int*   eidx     = (const int*)d_in[1];
  const int*   elab     = (const int*)d_in[2];
  /* d_in[3] batch: implied by contiguous equal-sized graphs */
  const float* node_emb = (const float*)d_in[4];
  const float* edge_emb = (const float*)d_in[5];
  const float* gat_W    = (const float*)d_in[6];
  const float* gat_We   = (const float*)d_in[7];
  const float* gat_as   = (const float*)d_in[8];
  const float* gat_ad   = (const float*)d_in[9];
  const float* gat_ae   = (const float*)d_in[10];
  const float* gat_b    = (const float*)d_in[11];
  const float* ln_w     = (const float*)d_in[12];
  const float* ln_b     = (const float*)d_in[13];
  const float* pl_w     = (const float*)d_in[14];
  const float* pl_b     = (const float*)d_in[15];
  const float* att_w    = (const float*)d_in[16];
  const float* att_b    = (const float*)d_in[17];
  const float* le1_w    = (const float*)d_in[18];
  const float* le1_b    = (const float*)d_in[19];
  const float* le2_w    = (const float*)d_in[20];
  const float* le3_w    = (const float*)d_in[21];
  const float* le3_b    = (const float*)d_in[22];
  const int* srcp = eidx;
  const int* dstp = eidx + NE;

  // -------- workspace layout --------
  char* ws = (char*)d_ws;
  size_t off = 0;
  auto alloc = [&](size_t bytes) { size_t r = off; off += (bytes + 255) & ~(size_t)255; return r; };
  float* bufX  = (float*)(ws + alloc((size_t)NN * DD * 4));   // node features x
  float* bufH  = (float*)(ws + alloc((size_t)NN * DD * 4));   // h = x@W ; later xq (float)
  float* bufO  = (float*)(ws + alloc((size_t)NN * DD * 4));   // GAT out ; later xq_lin
  float* xnew  = (float*)(ws + alloc((size_t)NN * DD * 4));
  int*   xqord = (int*)  (ws + alloc((size_t)NN * DD * 4));
  u16*   xpk   = (u16*)  (ws + alloc((size_t)NN * DD * 2));   // packed A fragments
  float* lacc  = (float*)(ws + alloc((size_t)NN * EDIM * 4)); // self-loop edge attrs
  float* degb  = (float*)(ws + alloc((size_t)NN * 4));
  float* as_n  = (float*)(ws + alloc((size_t)NN * NHD * 4));
  float* ad_n  = (float*)(ws + alloc((size_t)NN * NHD * 4));
  float* abuf  = (float*)(ws + alloc((size_t)NEF * NHD * 4));
  int*   mord  = (int*)  (ws + alloc((size_t)NN * NHD * 4));
  float* zbuf  = (float*)(ws + alloc((size_t)NN * NHD * 4));
  float* stats = (float*)(ws + alloc(4096));
  u16*   wpk   = (u16*)  (ws + alloc((size_t)NLAY * DD * DD * 2));
  u16*   ppk   = (u16*)  (ws + alloc((size_t)DD * DD * 2));
  float* weae  = (float*)(ws + alloc((size_t)NLAY * EDIM * NHD * 4));
  float* q1    = (float*)(ws + alloc((size_t)NN * 4));
  float* q2    = (float*)(ws + alloc((size_t)NN * 4));
  float* sbuf  = (float*)(ws + alloc((size_t)NEF * 4));
  int*   smord = (int*)  (ws + alloc((size_t)NN * 4));
  float* szb   = (float*)(ws + alloc((size_t)NN * 4));
  float* fa    = (float*)(ws + alloc((size_t)NN * 4));
  float* fb    = (float*)(ws + alloc((size_t)NN * 4));
  float* ft    = (float*)(ws + alloc((size_t)NN * 4));
  float* msg   = (float*)(ws + alloc((size_t)NN * 4));
  float* fit   = (float*)(ws + alloc((size_t)NN * 4));
  float* vals  = (float*)(ws + alloc((size_t)NBG * KTOP * 4));
  int*   perm  = (int*)  (ws + alloc((size_t)NBG * KTOP * 4));
  float* outp  = (float*)d_out;

  const int B = 256;
  const int Mtiles = NN / 16, Ntiles = DD / 16;             // 4096 x 8 waves
  const int nWaves = Mtiles * Ntiles;

  // -------- input embedding + self-loop edge attrs --------
  ge_pe_embed<<<blks((long long)NN * 64, B), B, 0, stream>>>(label, node_emb, bufX);
  ge_fill_f32<<<blks((long long)NN * EDIM, B), B, 0, stream>>>(lacc, NN * EDIM, 0.f);
  ge_fill_f32<<<blks(NN, B), B, 0, stream>>>(degb, NN, 0.f);
  ge_loop_attr_acc<<<blks((long long)NE * 4, B), B, 0, stream>>>(dstp, elab, edge_emb, lacc, degb);
  ge_loop_attr_div<<<blks((long long)NN * EDIM, B), B, 0, stream>>>(lacc, degb);

  // -------- pack all weights (bf16 B-fragments) + edge-attn folded weights --------
  for (int l = 0; l < NLAY; ++l) {
    ge_pack_b<<<blks(Ntiles * 4 * 32, B), B, 0, stream>>>(gat_W + (size_t)l * DD * DD,
                                                          wpk + (size_t)l * DD * DD, Ntiles, DD, DD);
    ge_make_weae<<<blks(EDIM * NHD, B), B, 0, stream>>>(gat_We + (size_t)l * EDIM * DD,
                                                        gat_ae + (size_t)l * NHD * CC,
                                                        weae + (size_t)l * EDIM * NHD);
  }
  ge_pack_b<<<blks(Ntiles * 4 * 32, B), B, 0, stream>>>(pl_w, ppk, Ntiles, DD, DD);

  // -------- 3 x (GATConv + graph LayerNorm + relu + residual) --------
  for (int l = 0; l < NLAY; ++l) {
    // h = x @ W[l]  (bf16 WMMA, f32 accumulate)
    ge_pack_a<<<blks((long long)Mtiles * 4 * 32, B), B, 0, stream>>>(bufX, xpk, Mtiles, DD);
    ge_gemm<4><<<blks((long long)nWaves * 32, B), B, 0, stream>>>(
        xpk, wpk + (size_t)l * DD * DD, (const float*)nullptr, bufH, DD, Ntiles, nWaves);
    // attention scalars
    ge_alpha_nodes<<<blks((long long)NN * NHD, B), B, 0, stream>>>(
        bufH, gat_as + (size_t)l * NHD * CC, gat_ad + (size_t)l * NHD * CC, as_n, ad_n);
    ge_fill_i32<<<blks((long long)NN * NHD, B), B, 0, stream>>>(mord, NN * NHD, NEG_ORD);
    ge_fill_f32<<<blks((long long)NN * NHD, B), B, 0, stream>>>(zbuf, NN * NHD, 0.f);
    ge_edge_score<<<blks(NEF, B), B, 0, stream>>>(srcp, dstp, elab, edge_emb, lacc,
                                                  weae + (size_t)l * EDIM * NHD, as_n, ad_n,
                                                  abuf, mord);
    ge_edge_exp<<<blks((long long)NEF * NHD, B), B, 0, stream>>>(srcp, dstp, abuf, mord, zbuf);
    // out = gat_b[l] + scatter(h[src] * alpha)
    ge_init_bias_rows<<<blks((long long)NN * DD, B), B, 0, stream>>>(bufO, gat_b + (size_t)l * DD);
    ge_edge_agg<<<blks((long long)NEF * NHD, B), B, 0, stream>>>(srcp, dstp, bufH, abuf, zbuf, bufO);
    // graph LayerNorm + relu + residual -> bufX
    ge_ln_stats<<<NBG, 256, 0, stream>>>(bufO, stats);
    ge_ln_apply<<<blks((long long)NN * DD, B), B, 0, stream>>>(bufO, stats,
                                                               ln_w + (size_t)l * DD,
                                                               ln_b + (size_t)l * DD, bufX);
  }

  // -------- ASAPooling --------
  ge_fill_i32<<<blks((long long)NN * DD, B), B, 0, stream>>>(xqord, NN * DD, NEG_ORD);
  ge_pool_max<<<blks((long long)NEF * NHD, B), B, 0, stream>>>(srcp, dstp, bufX, xqord);
  ge_ord2f<<<blks((long long)NN * DD, B), B, 0, stream>>>(xqord, bufH, NN * DD);
  // xq_lin = xq @ pool_lin_w + pool_lin_b  (bf16 WMMA)
  ge_pack_a<<<blks((long long)Mtiles * 4 * 32, B), B, 0, stream>>>(bufH, xpk, Mtiles, DD);
  ge_gemm<4><<<blks((long long)nWaves * 32, B), B, 0, stream>>>(xpk, ppk, pl_b, bufO, DD,
                                                                Ntiles, nWaves);
  ge_pool_q<<<blks(NN, B), B, 0, stream>>>(bufO, bufX, att_w, q1, q2);
  ge_fill_i32<<<blks(NN, B), B, 0, stream>>>(smord, NN, NEG_ORD);
  ge_fill_f32<<<blks(NN, B), B, 0, stream>>>(szb, NN, 0.f);
  ge_pool_score<<<blks(NEF, B), B, 0, stream>>>(srcp, dstp, q1, q2, att_b, sbuf, smord);
  ge_pool_exp<<<blks(NEF, B), B, 0, stream>>>(srcp, dstp, sbuf, smord, szb);
  ge_fill_f32<<<blks((long long)NN * DD, B), B, 0, stream>>>(xnew, NN * DD, 0.f);
  ge_pool_agg<<<blks((long long)NEF * NHD, B), B, 0, stream>>>(srcp, dstp, bufX, sbuf, szb, xnew);

  // -------- LEConv fitness + sigmoid --------
  ge_fit_node<<<blks(NN, B), B, 0, stream>>>(xnew, le1_w, le1_b, le2_w, le3_w, fa, fb, ft);
  ge_fill_f32<<<blks(NN, B), B, 0, stream>>>(msg, NN, 0.f);
  ge_fit_edge<<<blks(NEF, B), B, 0, stream>>>(srcp, dstp, fa, fb, msg);
  ge_fit_final<<<blks(NN, B), B, 0, stream>>>(msg, ft, le3_b, fit);

  // -------- per-graph top-k, gather, outputs --------
  ge_topk<<<NBG, NPGC, 0, stream>>>(fit, vals, perm);
  ge_gather_out<<<blks((long long)NBG * KTOP * DD, B), B, 0, stream>>>(xnew, perm, vals, outp);
  ge_batch_out<<<blks(NBG * KTOP, B), B, 0, stream>>>(outp);
}
